// VectorQuantize_36799279792263
// MI455X (gfx1250) — compile-verified, matrix operand order checked by reading, not run on hardware
//
#include <hip/hip_runtime.h>
#include <hip/hip_bf16.h>

// ---------------------------------------------------------------------------
// VQ-VAE vector quantization for MI455X (gfx1250).
// Dominant cost: 16384 x 8192 x 256 distance GEMM (~68.7 GFLOP) -> bf16 WMMA.
// argmin(||x-c||^2) == argmin(||c||^2 - 2 x.c): ||x||^2 dropped (row const).
// x.c via v_wmma_f32_16x16x32_bf16; everything value-bearing stays fp32.
// B panel staged via async global->LDS (ASYNCcnt) with double buffering.
// ---------------------------------------------------------------------------

typedef __attribute__((ext_vector_type(16))) __bf16 v16bf;
typedef __attribute__((ext_vector_type(8)))  float  v8f;
typedef __attribute__((ext_vector_type(4)))  int    v4i;

#define NROWS   16384      // B*L = 8*2048
#define DDIM    256
#define KCODES  8192
#define RTILES  1024       // NROWS/16
#define CTILES  512        // KCODES/16
#define KSTEPS  8          // DDIM/32
#define NCHUNK  (CTILES / 4)

// workspace byte offsets
#define WS_APACK  0u                          // bf16, NROWS*256   = 8 MB
#define WS_BPACK  (8u*1024u*1024u)            // bf16, KCODES*256  = 4 MB
#define WS_CNORM  (12u*1024u*1024u)           // f32, 8192         = 32 KB
#define WS_IDX    (WS_CNORM + 32u*1024u)      // i32, 16384        = 64 KB
#define WS_CSIZE  (WS_IDX + 64u*1024u)        // f32, 8192         = 32 KB
#define WS_CSUM   (WS_CSIZE + 32u*1024u)      // f32, 8192*256     = 8 MB

// output float offsets (tuple concatenated in return order)
#define OUT_Q      0u
#define OUT_IDX    4194304u
#define OUT_COMMIT 4210688u
#define OUT_NCB    4210689u
#define OUT_NECS   6307841u
#define OUT_NEV    6316033u

#if __has_builtin(__builtin_amdgcn_global_load_async_to_lds_b128) && \
    __has_builtin(__builtin_amdgcn_s_wait_asynccnt)
#define USE_ASYNC_LDS 1
#else
#define USE_ASYNC_LDS 0
#endif

#if USE_ASYNC_LDS
typedef __attribute__((address_space(1))) v4i* gptr128;
typedef __attribute__((address_space(3))) v4i* lptr128;
__device__ __forceinline__ void async_copy16(const void* g, void* l) {
    // per-lane 16B global -> LDS, tracked by ASYNCcnt
    __builtin_amdgcn_global_load_async_to_lds_b128((gptr128)g, (lptr128)l, 0, 0);
}
#endif

// k-offset inside a 16x32 bf16 A/B WMMA fragment (ISA 7.12.2 layout):
// lanes 0-15: VGPR0-3 hold K0..7 pairs, VGPR4-7 hold K16..23; lanes 16-31: +8
__device__ __forceinline__ int kofs(int j, int e, int h) {
    return ((j < 4) ? 0 : 16) + h * 8 + 2 * (j & 3) + e;
}

// --- pack x (B,D,L) -> A fragments, bf16, one wave per 16x32 tile ----------
__global__ void __launch_bounds__(256) vq_pack_a(const float* __restrict__ x,
                                                 __bf16* __restrict__ apack) {
    int wave = threadIdx.x >> 5, lane = threadIdx.x & 31;
    int tile = blockIdx.x * 8 + wave;           // RTILES*KSTEPS = 8192 tiles
    int rt = tile >> 3, ks = tile & 7;
    int m = lane & 15, h = lane >> 4;
    int n = rt * 16 + m;                        // flat row = b*2048 + l
    int b = n >> 11, l = n & 2047;
    const float* xb = x + (size_t)b * (DDIM * 2048) + l;
    __bf16* dst = apack + ((size_t)tile * 32 + lane) * 16;
#pragma unroll
    for (int j = 0; j < 8; ++j)
#pragma unroll
        for (int e = 0; e < 2; ++e) {
            int d = ks * 32 + kofs(j, e, h);
            dst[j * 2 + e] = (__bf16)xb[(size_t)d * 2048];
        }
}

// --- pack codebook (K,D) -> B fragments, bf16 ------------------------------
__global__ void __launch_bounds__(256) vq_pack_b(const float* __restrict__ cb,
                                                 __bf16* __restrict__ bpack) {
    int wave = threadIdx.x >> 5, lane = threadIdx.x & 31;
    int tile = blockIdx.x * 8 + wave;           // CTILES*KSTEPS = 4096 tiles
    int ct = tile >> 3, ks = tile & 7;
    int h = lane >> 4;
    int nc = ct * 16 + (lane & 15);             // codebook row = GEMM column
    const float* cr = cb + (size_t)nc * DDIM;
    __bf16* dst = bpack + ((size_t)tile * 32 + lane) * 16;
#pragma unroll
    for (int j = 0; j < 8; ++j)
#pragma unroll
        for (int e = 0; e < 2; ++e) {
            int d = ks * 32 + kofs(j, e, h);
            dst[j * 2 + e] = (__bf16)cr[d];
        }
}

// --- exact fp32 ||c||^2 per codebook row -----------------------------------
__global__ void __launch_bounds__(256) vq_cnorm(const float* __restrict__ cb,
                                                float* __restrict__ cnorm) {
    int wave = threadIdx.x >> 5, lane = threadIdx.x & 31;
    int row = blockIdx.x * 8 + wave;
    const float* cr = cb + (size_t)row * DDIM;
    float s = 0.f;
#pragma unroll
    for (int i = 0; i < 8; ++i) { float v = cr[lane + 32 * i]; s += v * v; }
#pragma unroll
    for (int off = 16; off; off >>= 1) s += __shfl_xor(s, off, 32);
    if (lane == 0) cnorm[row] = s;
}

// 4 column-tiles (64 columns) x 8 k-steps of WMMA + running argmin update
__device__ __forceinline__ void compute_chunk(
    const __bf16* __restrict__ bs, int chunk, int lane, const v16bf a[KSTEPS],
    const float* __restrict__ cnorm, float bestv[8], int bestc[8]) {
#pragma unroll
    for (int ctl = 0; ctl < 4; ++ctl) {
        int col = (chunk * 4 + ctl) * 16 + (lane & 15);
        float cn = cnorm[col];
        v8f acc = {0.f, 0.f, 0.f, 0.f, 0.f, 0.f, 0.f, 0.f};
#pragma unroll
        for (int ks = 0; ks < KSTEPS; ++ks) {
            const v16bf b =
                *(const v16bf*)(bs + ((ctl * KSTEPS + ks) * 32 + lane) * 16);
            acc = __builtin_amdgcn_wmma_f32_16x16x32_bf16(
                false, a[ks], false, b, (short)0, acc, false, false);
        }
        // C layout: VGPR r, lanes 0-15 -> row r, lanes 16-31 -> row r+8,
        // column = col for every r of this lane.
#pragma unroll
        for (int r = 0; r < 8; ++r) {
            float dv = cn - 2.0f * acc[r];
            if (dv < bestv[r]) { bestv[r] = dv; bestc[r] = col; }
        }
    }
}

// --- main kernel: WMMA distance GEMM fused with running argmin -------------
// 128 blocks x 256 threads (8 waves). Wave w owns row-tile rt = bid*8+w and
// holds its full 16x256 A panel in registers (8 x v16bf). B streamed through
// double-buffered 32 KB LDS chunks (4 column-tiles) shared by all 8 waves,
// staged with GLOBAL_LOAD_ASYNC_TO_LDS_B128 overlapping the WMMA stream.
__global__ void __launch_bounds__(256) vq_argmin(
    const __bf16* __restrict__ apack, const __bf16* __restrict__ bpack,
    const float* __restrict__ cnorm, int* __restrict__ idx_int,
    float* __restrict__ idx_f) {
    __shared__ __align__(16) __bf16 Bs[2][4 * KSTEPS * 32 * 16];  // 2 x 32 KB
    int tid = threadIdx.x, lane = tid & 31, wave = tid >> 5;
    int rt = blockIdx.x * 8 + wave;

    v16bf a[KSTEPS];
#pragma unroll
    for (int ks = 0; ks < KSTEPS; ++ks)
        a[ks] = *(const v16bf*)(apack + ((size_t)(rt * KSTEPS + ks) * 32 + lane) * 16);

    float bestv[8]; int bestc[8];
#pragma unroll
    for (int r = 0; r < 8; ++r) { bestv[r] = 3.4e38f; bestc[r] = 0; }

#if USE_ASYNC_LDS
    {   // prologue: stage chunk 0 into buffer 0
        const uint4* src = (const uint4*)bpack;
        uint4* dst = (uint4*)&Bs[0][0];
#pragma unroll
        for (int i = 0; i < 8; ++i)
            async_copy16(src + tid + 256 * i, dst + tid + 256 * i);
        __builtin_amdgcn_s_wait_asynccnt(0);
    }
    __syncthreads();
    for (int chunk = 0; chunk < NCHUNK; ++chunk) {
        int cur = chunk & 1;
        if (chunk + 1 < NCHUNK) {   // kick off next chunk before computing
            const uint4* src = (const uint4*)bpack + (size_t)(chunk + 1) * 2048;
            uint4* dst = (uint4*)&Bs[cur ^ 1][0];
#pragma unroll
            for (int i = 0; i < 8; ++i)
                async_copy16(src + tid + 256 * i, dst + tid + 256 * i);
        }
        compute_chunk(&Bs[cur][0], chunk, lane, a, cnorm, bestv, bestc);
        __builtin_amdgcn_s_wait_asynccnt(0);
        __syncthreads();
    }
#else
    for (int chunk = 0; chunk < NCHUNK; ++chunk) {
        __syncthreads();
        const uint4* src = (const uint4*)bpack + (size_t)chunk * 2048;
        uint4* dst = (uint4*)&Bs[0][0];
#pragma unroll
        for (int i = 0; i < 8; ++i) dst[tid + 256 * i] = src[tid + 256 * i];
        if (chunk + 1 < NCHUNK)     // global_prefetch_b8 for next chunk
            __builtin_prefetch((const void*)((const uint4*)bpack +
                               (size_t)(chunk + 1) * 2048 + tid * 8), 0, 1);
        __syncthreads();
        compute_chunk(&Bs[0][0], chunk, lane, a, cnorm, bestv, bestc);
    }
#endif

    // reduce argmin across the 16 lanes of each half-wave (prefer lower idx)
#pragma unroll
    for (int r = 0; r < 8; ++r) {
        float v = bestv[r]; int c = bestc[r];
#pragma unroll
        for (int off = 8; off; off >>= 1) {
            float ov = __shfl_xor(v, off, 16);
            int   oc = __shfl_xor(c, off, 16);
            if (ov < v || (ov == v && oc < c)) { v = ov; c = oc; }
        }
        if ((lane & 15) == 0) {
            int row = rt * 16 + r + (lane >> 4) * 8;
            idx_int[row] = c;
            idx_f[row] = (float)c;
        }
    }
}

// --- zero scratch accumulators + commit slot -------------------------------
__global__ void vq_zero(float* __restrict__ cs, float* __restrict__ csum,
                        float* __restrict__ commit) {
    size_t g = (size_t)blockIdx.x * 256 + threadIdx.x;
    if (g < KCODES) cs[g] = 0.f;
    if (g < (size_t)KCODES * DDIM) csum[g] = 0.f;
    if (g == 0) *commit = 0.f;
}

// --- gather quantized output, commit loss, segment-sum of x ----------------
// block = (b, 64-wide l-slice); 4 thread groups of 64 share d-ranges so all
// x/q traffic is coalesced along L. Codebook gathers hit L2 (8 MB resident).
__global__ void __launch_bounds__(256) vq_gather(
    const float* __restrict__ x, const float* __restrict__ cb,
    const int* __restrict__ idx, float* __restrict__ outq,
    float* __restrict__ csum, float* __restrict__ commit) {
    __shared__ float red[256];
    int bid = blockIdx.x;                    // 256 blocks
    int b = bid >> 5;
    int l = (bid & 31) * 64 + (threadIdx.x & 63);
    int dg = threadIdx.x >> 6;
    int n = b * 2048 + l;
    int k = idx[n];
    const float* crow = cb + (size_t)k * DDIM;
    const float* xb = x + (size_t)b * (DDIM * 2048) + l;
    float* qb = outq + (size_t)b * (DDIM * 2048) + l;
    float acc = 0.f;
#pragma unroll 4
    for (int j = 0; j < 64; ++j) {
        int d = dg * 64 + j;
        float q = crow[d];
        float xi = xb[(size_t)d * 2048];
        float t = q - xi;
        qb[(size_t)d * 2048] = xi + t;       // STE: x + (q - x), bit-matching
        acc += t * t;
        atomicAdd(&csum[(size_t)k * DDIM + d], xi);
    }
    red[threadIdx.x] = acc;
    __syncthreads();
    for (int s = 128; s; s >>= 1) {
        if (threadIdx.x < s) red[threadIdx.x] += red[threadIdx.x + s];
        __syncthreads();
    }
    if (threadIdx.x == 0)
        atomicAdd(commit, red[0] * (1.0f / (float)((size_t)NROWS * DDIM)));
}

// --- cluster size histogram ------------------------------------------------
__global__ void vq_count(const int* __restrict__ idx, float* __restrict__ cs) {
    int n = blockIdx.x * 256 + threadIdx.x;  // 64 blocks
    atomicAdd(&cs[idx[n]], 1.0f);
}

// --- EMA finalize ----------------------------------------------------------
__global__ void vq_finalize(const float* __restrict__ ema_val,
                            const float* __restrict__ ema_cs,
                            const float* __restrict__ csum,
                            const float* __restrict__ cs,
                            float* __restrict__ ncb, float* __restrict__ necs,
                            float* __restrict__ nev) {
    int g = blockIdx.x * 256 + threadIdx.x;  // 8192 blocks -> 2M elems
    int k = g >> 8, d = g & 255;
    float c = ema_cs[k] * 0.8f + cs[k];
    float v = ema_val[g] * 0.8f + csum[g];
    nev[g] = v;
    ncb[g] = v / (c + 1e-5f);
    if (d == 0) necs[k] = c;
}

extern "C" void kernel_launch(void* const* d_in, const int* in_sizes, int n_in,
                              void* d_out, int out_size, void* d_ws,
                              size_t ws_size, hipStream_t stream) {
    const float* x       = (const float*)d_in[0];   // (8,256,2048)
    const float* cb      = (const float*)d_in[1];   // (8192,256)
    const float* ema_val = (const float*)d_in[2];   // (8192,256)
    const float* ema_cs  = (const float*)d_in[3];   // (8192,)
    float* out = (float*)d_out;

    char* ws = (char*)d_ws;
    __bf16* apack = (__bf16*)(ws + WS_APACK);
    __bf16* bpack = (__bf16*)(ws + WS_BPACK);
    float*  cnorm = (float*)(ws + WS_CNORM);
    int*    idxw  = (int*)(ws + WS_IDX);
    float*  cs    = (float*)(ws + WS_CSIZE);
    float*  csum  = (float*)(ws + WS_CSUM);

    vq_zero<<<8192, 256, 0, stream>>>(cs, csum, out + OUT_COMMIT);
    vq_pack_a<<<1024, 256, 0, stream>>>(x, apack);
    vq_pack_b<<<512, 256, 0, stream>>>(cb, bpack);
    vq_cnorm<<<1024, 256, 0, stream>>>(cb, cnorm);
    vq_argmin<<<128, 256, 0, stream>>>(apack, bpack, cnorm, idxw, out + OUT_IDX);
    vq_gather<<<256, 256, 0, stream>>>(x, cb, idxw, out + OUT_Q, csum,
                                       out + OUT_COMMIT);
    vq_count<<<64, 256, 0, stream>>>(idxw, cs);
    vq_finalize<<<8192, 256, 0, stream>>>(ema_val, ema_cs, csum, cs,
                                          out + OUT_NCB, out + OUT_NECS,
                                          out + OUT_NEV);
}
